// WordGenerator_18906446037056
// MI455X (gfx1250) — matrix operand order repaired and checked by reading, not compiled, not run on hardware
//
#include <hip/hip_runtime.h>
#include <hip/hip_bf16.h>

// Bi-LSTM word generator for MI455X (gfx1250, wave32, WMMA).
// Plan: precompute all input-side projections as large parallel WMMA GEMMs,
// keep only the h@Whh^T recurrences in the per-step sequential chain.
// Recurrent weights (reused 128x) are pre-converted to f16 once; one-shot
// weights are converted on the fly. All matmuls use v_wmma_f32_16x16x32_f16.

#define SEQ   128
#define BATCH 64
#define HID   512
#define VOCAB 32000

typedef __attribute__((ext_vector_type(16))) _Float16 v16h;
typedef __attribute__((ext_vector_type(8)))  _Float16 h8;
typedef __attribute__((ext_vector_type(8)))  float    v8f;

__device__ __forceinline__ float sigf(float x) { return 1.0f / (1.0f + __expf(-x)); }

// ---------------------------------------------------------------------------
// Embedding gather (padding_idx = 0), fp32 -> f16, laid out as [S*B, H] rows.
// reshape(S,B,H) of E[x] keeps linear order, so row r of the [S,B,H] view is
// simply E[x_flat[r]].
// ---------------------------------------------------------------------------
__global__ __launch_bounds__(256) void k_embed(const int* __restrict__ x,
                                               const float* __restrict__ emb,
                                               _Float16* __restrict__ out) {
  int gid = blockIdx.x * 256 + threadIdx.x;       // SEQ*BATCH*HID total
  int r   = gid >> 9;                             // /HID
  int h   = gid & (HID - 1);
  int tok = x[r];
  float v = (tok == 0) ? 0.0f : emb[(size_t)tok * HID + h];
  out[gid] = (_Float16)v;
}

// ---------------------------------------------------------------------------
// One-shot fp32 -> f16 weight conversion (for weights reused across steps).
// ---------------------------------------------------------------------------
__global__ __launch_bounds__(256) void k_cvt(const float* __restrict__ src,
                                             _Float16* __restrict__ dst) {
  int gid = blockIdx.x * 256 + threadIdx.x;
  dst[gid] = (_Float16)src[gid];
}

// ---------------------------------------------------------------------------
// Initial-state staging: h-states to f16, c-states copied fp32.
// ---------------------------------------------------------------------------
__global__ __launch_bounds__(256) void k_init(const float* __restrict__ h0f,
                                              const float* __restrict__ c0f,
                                              const float* __restrict__ h0b,
                                              const float* __restrict__ c0b,
                                              const float* __restrict__ h01,
                                              const float* __restrict__ h02,
                                              const float* __restrict__ c02,
                                              _Float16* __restrict__ hf,
                                              _Float16* __restrict__ hb,
                                              _Float16* __restrict__ h1,
                                              _Float16* __restrict__ h2,
                                              float* __restrict__ cf,
                                              float* __restrict__ cb,
                                              float* __restrict__ c2) {
  int gid = blockIdx.x * 256 + threadIdx.x;       // BATCH*2*HID total
  if (gid < BATCH * HID) {
    hf[gid] = (_Float16)h0f[gid];
    hb[gid] = (_Float16)h0b[gid];
    cf[gid] = c0f[gid];
    cb[gid] = c0b[gid];
  }
  h1[gid] = (_Float16)h01[gid];
  h2[gid] = (_Float16)h02[gid];
  c2[gid] = c02[gid];
}

// ---------------------------------------------------------------------------
// Per-lane fragment construction uses contiguous k order for both A and B:
//   A (16x32 f16): lane L: M = L&15; halves 0..15 -> K = 16*(L>>4)+h.
//   B (32x16 f16): lane L: N = L&15; halves 0..15 -> K = 16*(L>>4)+h.
//   C/D (16x16 f32): lane L, reg r: M = 8*(L>>4)+r, N = L&15.
// Block = 128 threads = 4 waves; wave w owns the 16x16 tile at
// (m0 = 64*by + 16*w, n0 = 16*bx).  M % 64 == 0, N % 16 == 0, K % 32 == 0.
// ---------------------------------------------------------------------------
__device__ __forceinline__ v16h load_a_frag(const _Float16* arow, int kk) {
  const h8* ap = (const h8*)(arow + kk);          // 16B-aligned (lda % 8 == 0)
  h8 alo = ap[0];                                 // halves 0..7
  h8 ahi = ap[1];                                 // halves 8..15 (k + 8)
  return __builtin_shufflevector(alo, ahi,
                                 0, 1, 2, 3, 4, 5, 6, 7,
                                 8, 9, 10, 11, 12, 13, 14, 15);
}

// WMMA GEMM, fp32 weights converted per-fragment (one-shot weights):
//   Cout[M,N] = A[M,K](f16, stride lda) @ W[N,K]^T(f32) (+Cin) (+bias)
__global__ __launch_bounds__(128) void k_gemm(const _Float16* __restrict__ A, int lda,
                                              const float* __restrict__ W, int K,
                                              const float* __restrict__ bias,
                                              const float* __restrict__ Cin,
                                              float* __restrict__ Cout, int ldc) {
  const int lane = threadIdx.x & 31;
  const int wave = threadIdx.x >> 5;
  const int lm   = lane & 15;
  const int hk   = lane >> 4;                     // 0 or 1 (lane half)
  const int m0   = blockIdx.y * 64 + wave * 16;
  const int n0   = blockIdx.x * 16;

  v8f acc = {};
  if (Cin) {
#pragma unroll
    for (int r = 0; r < 8; ++r)
      acc[r] = Cin[(size_t)(m0 + 8 * hk + r) * ldc + n0 + lm];
  }

  const _Float16* arow = A + (size_t)(m0 + lm) * lda + hk * 16;
  const float*    wrow = W + (size_t)(n0 + lm) * K + hk * 16;

  for (int kk = 0; kk < K; kk += 32) {
    v16h a = load_a_frag(arow, kk);
    v16h b;
    const float* wp = wrow + kk;
#pragma unroll
    for (int i = 0; i < 16; ++i) b[i] = (_Float16)wp[i];
    acc = __builtin_amdgcn_wmma_f32_16x16x32_f16(false, a, false, b,
                                                 (short)0, acc, false, false);
  }

  const float bv = bias ? bias[n0 + lm] : 0.0f;
#pragma unroll
  for (int r = 0; r < 8; ++r)
    Cout[(size_t)(m0 + 8 * hk + r) * ldc + n0 + lm] = acc[r] + bv;
}

// WMMA GEMM, pre-converted f16 weights (recurrent chain: no cvt in loop,
// half the L2 weight traffic):
//   Cout[M,N] = A[M,K](f16, stride lda) @ Wh[N,K]^T(f16) (+Cin)
__global__ __launch_bounds__(128) void k_gemm_h(const _Float16* __restrict__ A, int lda,
                                                const _Float16* __restrict__ Wh, int K,
                                                const float* __restrict__ Cin,
                                                float* __restrict__ Cout, int ldc) {
  const int lane = threadIdx.x & 31;
  const int wave = threadIdx.x >> 5;
  const int lm   = lane & 15;
  const int hk   = lane >> 4;
  const int m0   = blockIdx.y * 64 + wave * 16;
  const int n0   = blockIdx.x * 16;

  v8f acc = {};
  if (Cin) {
#pragma unroll
    for (int r = 0; r < 8; ++r)
      acc[r] = Cin[(size_t)(m0 + 8 * hk + r) * ldc + n0 + lm];
  }

  const _Float16* arow = A  + (size_t)(m0 + lm) * lda + hk * 16;
  const _Float16* wrow = Wh + (size_t)(n0 + lm) * K + hk * 16;   // 32B-aligned

  for (int kk = 0; kk < K; kk += 32) {
    v16h a = load_a_frag(arow, kk);
    v16h b = *(const v16h*)(wrow + kk);           // 16 contiguous halves
    acc = __builtin_amdgcn_wmma_f32_16x16x32_f16(false, a, false, b,
                                                 (short)0, acc, false, false);
  }

#pragma unroll
  for (int r = 0; r < 8; ++r)
    Cout[(size_t)(m0 + 8 * hk + r) * ldc + n0 + lm] = acc[r];
}

// ---------------------------------------------------------------------------
// LSTM pointwise: gates[B,4H] (pre-activation, no biases) + bih + bhh,
// PyTorch gate order i,f,g,o.  c' = sig(f)*c + sig(i)*tanh(g);
// h' = sig(o)*tanh(c').  c_out may be null (cell-1 discards its c).
// h_out: dense f16 [B,H].  h_out2 (optional): f16 with row stride 2*HID
// (used to write fwd/bwd h into the concat xin buffer).
// ---------------------------------------------------------------------------
__global__ __launch_bounds__(256) void k_lstm_pw(const float* __restrict__ gates,
                                                 const float* __restrict__ bih,
                                                 const float* __restrict__ bhh,
                                                 const float* __restrict__ c_in,
                                                 float* __restrict__ c_out,
                                                 _Float16* __restrict__ h_out,
                                                 _Float16* __restrict__ h_out2,
                                                 int Hh) {
  int gid = blockIdx.x * 256 + threadIdx.x;       // BATCH*Hh total
  int b = gid / Hh;
  int j = gid - b * Hh;
  const float* gr = gates + (size_t)b * 4 * Hh;

  float gi = sigf(gr[j]           + bih[j]           + bhh[j]);
  float gf = sigf(gr[Hh + j]      + bih[Hh + j]      + bhh[Hh + j]);
  float gg = tanhf(gr[2 * Hh + j] + bih[2 * Hh + j]  + bhh[2 * Hh + j]);
  float go = sigf(gr[3 * Hh + j]  + bih[3 * Hh + j]  + bhh[3 * Hh + j]);

  float c = gf * c_in[gid] + gi * gg;
  float h = go * tanhf(c);

  if (c_out) c_out[gid] = c;
  h_out[gid] = (_Float16)h;
  if (h_out2) h_out2[(size_t)b * (2 * HID) + j] = (_Float16)h;
}

// ---------------------------------------------------------------------------
extern "C" void kernel_launch(void* const* d_in, const int* in_sizes, int n_in,
                              void* d_out, int out_size, void* d_ws, size_t ws_size,
                              hipStream_t stream) {
  (void)in_sizes; (void)n_in; (void)out_size; (void)ws_size;

  const int*   x     = (const int*)  d_in[0];
  const float* emb   = (const float*)d_in[1];
  const float* h0f   = (const float*)d_in[2];
  const float* c0f   = (const float*)d_in[3];
  const float* h0b   = (const float*)d_in[4];
  const float* c0b   = (const float*)d_in[5];
  const float* h01   = (const float*)d_in[6];
  const float* h02   = (const float*)d_in[7];
  const float* c02   = (const float*)d_in[8];
  const float* wih_f = (const float*)d_in[9];
  const float* whh_f = (const float*)d_in[10];
  const float* bih_f = (const float*)d_in[11];
  const float* bhh_f = (const float*)d_in[12];
  const float* wih_b = (const float*)d_in[13];
  const float* whh_b = (const float*)d_in[14];
  const float* bih_b = (const float*)d_in[15];
  const float* bhh_b = (const float*)d_in[16];
  const float* wih1  = (const float*)d_in[17];
  const float* whh1  = (const float*)d_in[18];
  const float* bih1  = (const float*)d_in[19];
  const float* bhh1  = (const float*)d_in[20];
  const float* wih2  = (const float*)d_in[21];
  const float* whh2  = (const float*)d_in[22];
  const float* bih2  = (const float*)d_in[23];
  const float* bhh2  = (const float*)d_in[24];
  const float* wlin  = (const float*)d_in[25];
  const float* blin  = (const float*)d_in[26];
  float* out = (float*)d_out;

  // ---- workspace carve (offsets 256B-aligned; Gf and Gb contiguous so the
  //      combined region is reused as G1 [S*B, 8H] after the fwd/bwd scans) --
  char* ws = (char*)d_ws;
  size_t off = 0;
  auto take = [&](size_t bytes) -> char* {
    char* p = ws + off;
    off = (off + bytes + 255) & ~(size_t)255;
    return p;
  };
  const size_t SB   = (size_t)SEQ * BATCH;                      // 8192
  const size_t WDIR = (size_t)4 * HID * HID;                    // 2048*512
  const size_t WSTK = (size_t)8 * HID * 2 * HID;                // 4096*1024
  _Float16* xemb   = (_Float16*)take(SB * HID * sizeof(_Float16));
  float*    Gf     = (float*)   take(SB * 4 * HID * sizeof(float));  // 64 MiB
  float*    Gb     = (float*)   take(SB * 4 * HID * sizeof(float));  // 64 MiB
  float*    G1     = Gf;                                             // reuse both
  _Float16* xin    = (_Float16*)take(SB * 2 * HID * sizeof(_Float16));
  float*    scr2   = (float*)   take((size_t)BATCH * 8 * HID * sizeof(float));
  _Float16* whhf_h = (_Float16*)take(WDIR * sizeof(_Float16));  // f16 recurrent W
  _Float16* whhb_h = (_Float16*)take(WDIR * sizeof(_Float16));
  _Float16* whh1_h = (_Float16*)take(WSTK * sizeof(_Float16));
  _Float16* wih2_h = (_Float16*)take(WSTK * sizeof(_Float16));
  _Float16* whh2_h = (_Float16*)take(WSTK * sizeof(_Float16));
  _Float16* hf     = (_Float16*)take((size_t)BATCH * HID * sizeof(_Float16));
  _Float16* hb     = (_Float16*)take((size_t)BATCH * HID * sizeof(_Float16));
  _Float16* h1     = (_Float16*)take((size_t)BATCH * 2 * HID * sizeof(_Float16));
  _Float16* h2     = (_Float16*)take((size_t)BATCH * 2 * HID * sizeof(_Float16));
  float*    cf     = (float*)   take((size_t)BATCH * HID * sizeof(float));
  float*    cb     = (float*)   take((size_t)BATCH * HID * sizeof(float));
  float*    c2     = (float*)   take((size_t)BATCH * 2 * HID * sizeof(float));

  // ---- prologue: embed, state staging, recurrent-weight f16 conversion ----
  k_embed<<<SB * HID / 256, 256, 0, stream>>>(x, emb, xemb);
  k_init<<<(BATCH * 2 * HID) / 256, 256, 0, stream>>>(h0f, c0f, h0b, c0b,
                                                      h01, h02, c02,
                                                      hf, hb, h1, h2, cf, cb, c2);
  k_cvt<<<WDIR / 256, 256, 0, stream>>>(whh_f, whhf_h);
  k_cvt<<<WDIR / 256, 256, 0, stream>>>(whh_b, whhb_h);
  k_cvt<<<WSTK / 256, 256, 0, stream>>>(whh1, whh1_h);
  k_cvt<<<WSTK / 256, 256, 0, stream>>>(wih2, wih2_h);
  k_cvt<<<WSTK / 256, 256, 0, stream>>>(whh2, whh2_h);

  // ---- batched input projections for fwd/bwd over all timesteps -----------
  // Gf/Gb[s*B + b, :] = xemb[row] @ Wih^T   (biases added in pointwise)
  dim3 gbig(4 * HID / 16, SB / 64);                             // (128,128)
  k_gemm<<<gbig, 128, 0, stream>>>(xemb, HID, wih_f, HID, nullptr, nullptr, Gf, 4 * HID);
  k_gemm<<<gbig, 128, 0, stream>>>(xemb, HID, wih_b, HID, nullptr, nullptr, Gb, 4 * HID);

  // ---- sequential fwd/bwd scans (only h@Whh^T is in the chain) ------------
  dim3 gdir(4 * HID / 16, 1);                                   // (128,1)
  for (int s = 0; s < SEQ; ++s) {
    float* gfs = Gf + (size_t)s * BATCH * 4 * HID;
    float* gbs = Gb + (size_t)(SEQ - 1 - s) * BATCH * 4 * HID;  // bwd walks reversed
    k_gemm_h<<<gdir, 128, 0, stream>>>(hf, HID, whhf_h, HID, gfs, gfs, 4 * HID);
    k_gemm_h<<<gdir, 128, 0, stream>>>(hb, HID, whhb_h, HID, gbs, gbs, 4 * HID);
    _Float16* xslot = xin + (size_t)s * BATCH * 2 * HID;
    k_lstm_pw<<<BATCH * HID / 256, 256, 0, stream>>>(gfs, bih_f, bhh_f, cf, cf, hf, xslot, HID);
    k_lstm_pw<<<BATCH * HID / 256, 256, 0, stream>>>(gbs, bih_b, bhh_b, cb, cb, hb, xslot + HID, HID);
  }

  // ---- batched input projection for stacked cell-1 ------------------------
  dim3 gG1(8 * HID / 16, SB / 64);                              // (256,128)
  k_gemm<<<gG1, 128, 0, stream>>>(xin, 2 * HID, wih1, 2 * HID, nullptr, nullptr, G1, 8 * HID);

  // ---- sequential stacked scan (cell-1 fed c2; its own c discarded) -------
  dim3 gstk(8 * HID / 16, 1);                                   // (256,1)
  for (int s = 0; s < SEQ; ++s) {
    float* g1s = G1 + (size_t)s * BATCH * 8 * HID;
    // cell-1: gates1 = G1[s] + h1 @ whh1^T
    k_gemm_h<<<gstk, 128, 0, stream>>>(h1, 2 * HID, whh1_h, 2 * HID, g1s, g1s, 8 * HID);
    k_lstm_pw<<<BATCH * 2 * HID / 256, 256, 0, stream>>>(g1s, bih1, bhh1, c2,
                                                         /*c_out*/nullptr, h1, nullptr, 2 * HID);
    // cell-2: gates2 = h1n @ wih2^T + h2 @ whh2^T
    k_gemm_h<<<gstk, 128, 0, stream>>>(h1, 2 * HID, wih2_h, 2 * HID, nullptr, scr2, 8 * HID);
    k_gemm_h<<<gstk, 128, 0, stream>>>(h2, 2 * HID, whh2_h, 2 * HID, scr2, scr2, 8 * HID);
    k_lstm_pw<<<BATCH * 2 * HID / 256, 256, 0, stream>>>(scr2, bih2, bhh2, c2, c2, h2, nullptr, 2 * HID);
  }

  // ---- final classifier: out = h2 @ wlin^T + blin, [64, 32000] ------------
  dim3 gfin(VOCAB / 16, 1);                                     // (2000,1)
  k_gemm<<<gfin, 128, 0, stream>>>(h2, 2 * HID, wlin, 2 * HID, blin, nullptr, out, VOCAB);
}
